// ARLSTMModel_65936337928698
// MI455X (gfx1250) — compile-verified, matrix operand order checked by reading, not run on hardware
//
#include <hip/hip_runtime.h>

// ---- problem constants (match reference) ----
#define HID   512            // H
#define DOUT  64             // D
#define BSZ   256            // BS
#define NTIME 1024           // NT
#define G4H   2048           // 4*H

// ---- CDNA5 WMMA types ----
typedef __attribute__((ext_vector_type(16))) __bf16 v16bf;
typedef __attribute__((ext_vector_type(8)))  float  v8f;

union Frag16 {
    v16bf v;
    uint4 q[2];
};

static __device__ __forceinline__ unsigned short f2bf(float x) {
    unsigned u = __float_as_uint(x);
    unsigned r = (u + 0x7fffu + ((u >> 16) & 1u)) >> 16;   // RNE
    return (unsigned short)r;
}

static __device__ __forceinline__ float fsigmoid(float x) {
    return 1.0f / (1.0f + __expf(-x));
}
static __device__ __forceinline__ float ftanh_fast(float x) {
    x = fminf(fmaxf(x, -15.0f), 15.0f);
    float e = __expf(2.0f * x);
    return (e - 1.0f) / (e + 1.0f);
}

// Load a 16-bit 16x32 WMMA A/B fragment from a row-major source.
// ISA layout (CDNA5 ISA, 16-bit A 16x32):
//   lanes 0-15 : V0..V3 = K 0..7 (pairs),  V4..V7 = K 16..23
//   lanes16-31 : V0..V3 = K 8..15,         V4..V7 = K 24..31
// Per lane: two contiguous 16-byte runs at +0 and +32 bytes from
// p = row + k0 + (laneHi ? 8 : 0) elements.
static __device__ __forceinline__ v16bf load_frag16(const unsigned short* rowk, int lane) {
    const unsigned short* p = rowk + ((lane & 16) ? 8 : 0);
    Frag16 f;
    f.q[0] = *(const uint4*)(p);        // K pairs 0..7   (b128)
    f.q[1] = *(const uint4*)(p + 16);   // K pairs 16..23 (b128)
    return f.v;
}

#define WMMA_BF16(A, B, C) \
    __builtin_amdgcn_wmma_f32_16x16x32_bf16(false, (A), false, (B), (short)0, (C), false, false)

// ---------------------------------------------------------------------------
// Prep: f32 weights -> bf16 in workspace; fold biases. Runs once per launch.
// ---------------------------------------------------------------------------
__global__ void arlstm_prep(const float* __restrict__ Wih,  const float* __restrict__ Whh,
                            const float* __restrict__ bih,  const float* __restrict__ bhh,
                            const float* __restrict__ Wproj,const float* __restrict__ bproj,
                            unsigned short* __restrict__ wih_b,
                            unsigned short* __restrict__ whh_b,
                            unsigned short* __restrict__ wproj_b,
                            float* __restrict__ bias_o, float* __restrict__ bproj_o)
{
    const int stride = gridDim.x * blockDim.x;
    const int i0 = blockIdx.x * blockDim.x + threadIdx.x;
    for (int i = i0; i < G4H * HID;   i += stride) whh_b[i]   = f2bf(Whh[i]);
    for (int i = i0; i < G4H * DOUT;  i += stride) wih_b[i]   = f2bf(Wih[i]);
    for (int i = i0; i < 2*DOUT*HID;  i += stride) wproj_b[i] = f2bf(Wproj[i]);
    for (int i = i0; i < G4H;         i += stride) bias_o[i]  = bih[i] + bhh[i];
    for (int i = i0; i < 2*DOUT;      i += stride) bproj_o[i] = bproj[i];
}

// ---------------------------------------------------------------------------
// Main: 16 persistent workgroups, one per 16-row batch tile. 512 threads =
// 16 wave32. Wave w owns hidden slices {w, w+16}: computes their i/f/g/o gate
// tiles via bf16 WMMA streaming W from L2 each step, keeps cell state c in
// D-layout registers, writes new h (bf16) to ping-pong LDS. Waves 0-7 do the
// projection. Batch chains are independent -> no grid sync, 3 barriers/step.
// ---------------------------------------------------------------------------
__global__ __launch_bounds__(512, 1) void arlstm_main(
    const float* __restrict__ h0,  const float* __restrict__ c0,
    const float* __restrict__ yt,  const float* __restrict__ eps,
    const unsigned short* __restrict__ wih_g,
    const unsigned short* __restrict__ whh_g,
    const unsigned short* __restrict__ wproj_g,
    const float* __restrict__ bias, const float* __restrict__ bproj,
    float* __restrict__ out)
{
    __shared__ unsigned short hbuf[2][16][HID];   // h, bf16, ping-pong (A operand)
    __shared__ unsigned short xbuf[2][16][DOUT];  // y feedback, bf16, ping-pong
    __shared__ float          projbuf[16][2*DOUT];// mu | logvar staging

    const int tid  = threadIdx.x;
    const int lane = tid & 31;
    const int wv   = tid >> 5;           // wave id 0..15
    const int l15  = lane & 15;
    const int lhi  = (lane >> 4) & 1;    // 0: rows 0-7, 1: rows 8-15 (D layout)
    const int m0   = blockIdx.x * 16;    // batch tile base row

    // ---- init LDS state (buffer 0) from f32 inputs ----
    for (int idx = tid; idx < 16 * HID; idx += 512) {
        int m = idx >> 9, k = idx & (HID - 1);
        hbuf[0][m][k] = f2bf(h0[(size_t)(m0 + m) * HID + k]);
    }
    for (int idx = tid; idx < 16 * DOUT; idx += 512) {
        int m = idx >> 6, j = idx & (DOUT - 1);
        xbuf[0][m][j] = f2bf(yt[((size_t)(m0 + m) * NTIME + 0) * DOUT + j]);
    }

    // gate-tile columns for this wave: [slice(2)][gate i,f,g,o]
    int ncol[8];
#pragma unroll
    for (int sl = 0; sl < 2; ++sl) {
        int s = wv + 16 * sl;
#pragma unroll
        for (int g = 0; g < 4; ++g) ncol[sl * 4 + g] = g * HID + 16 * s + l15;
    }

    // cell state in registers, D-matrix layout: (m,n)=(lhi*8+r, 16*s+l15)
    float creg[2][8];
#pragma unroll
    for (int sl = 0; sl < 2; ++sl) {
        int s = wv + 16 * sl;
#pragma unroll
        for (int r = 0; r < 8; ++r)
            creg[sl][r] = c0[(size_t)(m0 + lhi * 8 + r) * HID + 16 * s + l15];
    }
    __syncthreads();

    const size_t SZ = (size_t)BSZ * NTIME * DOUT;
    float* out_ys = out;
    float* out_mu = out + SZ;
    float* out_lv = out + 2 * SZ;

    for (int t = 0; t < NTIME; ++t) {
        const int rp = t & 1;        // read buffer
        const int wp = rp ^ 1;       // write buffer

        // Launder a zero *offset* (not the pointers!) through an SGPR so the
        // weight loads are loop-variant (no LICM hoist -> no scratch spills)
        // while the base pointers keep their inferred GLOBAL address space
        // (global_load_b128 saddr form, not flat_load with 64-bit vaddr).
        unsigned toff = 0;
        asm volatile("" : "+s"(toff));
        const unsigned short* wih   = wih_g   + toff;
        const unsigned short* whh   = whh_g   + toff;
        const unsigned short* wproj = wproj_g + toff;

        // ================= gates = [x|h] @ W^T + bias =================
        v8f acc[8];
#pragma unroll
        for (int tl = 0; tl < 8; ++tl) {
            float b = bias[ncol[tl]];
#pragma unroll
            for (int r = 0; r < 8; ++r) acc[tl][r] = b;
        }
        // x part: K = 64 (W_ih), A from xbuf[rp]
#pragma unroll
        for (int kk = 0; kk < 2; ++kk) {
            v16bf a = load_frag16(&xbuf[rp][l15][32 * kk], lane);
#pragma unroll
            for (int tl = 0; tl < 8; ++tl) {
                v16bf b = load_frag16(wih + ncol[tl] * DOUT + 32 * kk, lane);
                acc[tl] = WMMA_BF16(a, b, acc[tl]);
            }
        }
        // h part: K = 512 (W_hh), A from hbuf[rp]
#pragma unroll 4
        for (int kk = 0; kk < 16; ++kk) {
            v16bf a = load_frag16(&hbuf[rp][l15][32 * kk], lane);
#pragma unroll
            for (int tl = 0; tl < 8; ++tl) {
                v16bf b = load_frag16(whh + ncol[tl] * HID + 32 * kk, lane);
                acc[tl] = WMMA_BF16(a, b, acc[tl]);
            }
        }

        // ======= pointwise LSTM update; write new h straight to hbuf[wp] =======
        // (ping-pong: no WAR hazard against other waves still reading hbuf[rp])
#pragma unroll
        for (int sl = 0; sl < 2; ++sl) {
            int s = wv + 16 * sl;
#pragma unroll
            for (int r = 0; r < 8; ++r) {
                float ii = fsigmoid(acc[sl * 4 + 0][r]);
                float ff = fsigmoid(acc[sl * 4 + 1][r]);
                float gg = ftanh_fast(acc[sl * 4 + 2][r]);
                float oo = fsigmoid(acc[sl * 4 + 3][r]);
                float cc = ff * creg[sl][r] + ii * gg;
                creg[sl][r] = cc;
                hbuf[wp][lhi * 8 + r][16 * s + l15] = f2bf(oo * ftanh_fast(cc));
            }
        }
        __syncthreads();   // new h (hbuf[wp]) visible

        // ================= projection: [16,128] = h @ Wproj^T =================
        if (wv < 8) {
            int n = 16 * wv + l15;
            v8f pacc;
            float b = bproj[n];
#pragma unroll
            for (int r = 0; r < 8; ++r) pacc[r] = b;
#pragma unroll 4
            for (int kk = 0; kk < 16; ++kk) {
                v16bf a  = load_frag16(&hbuf[wp][l15][32 * kk], lane);
                v16bf bb = load_frag16(wproj + n * HID + 32 * kk, lane);
                pacc = WMMA_BF16(a, bb, pacc);
            }
#pragma unroll
            for (int r = 0; r < 8; ++r) projbuf[lhi * 8 + r][n] = pacc[r];
        }
        __syncthreads();   // proj visible

        // ================= reparameterized sample + outputs =================
        for (int e = tid; e < 16 * DOUT; e += 512) {
            int m = e >> 6, j = e & (DOUT - 1);
            float mu = projbuf[m][j];
            float lv = projbuf[m][DOUT + j];
            size_t gidx = ((size_t)(m0 + m) * NTIME + t) * DOUT + j;
            float ep = __builtin_nontemporal_load(&eps[gidx]);
            float y  = ep * __expf(0.5f * lv) + mu;
            __builtin_nontemporal_store(y,  &out_ys[gidx]);
            __builtin_nontemporal_store(mu, &out_mu[gidx]);
            __builtin_nontemporal_store(lv, &out_lv[gidx]);
            xbuf[wp][m][j] = f2bf(y);   // feedback for step t+1 (its read buffer)
        }
        __syncthreads();   // xbuf[wp] visible for next step's gates
    }
}

// ---------------------------------------------------------------------------
extern "C" void kernel_launch(void* const* d_in, const int* in_sizes, int n_in,
                              void* d_out, int out_size, void* d_ws, size_t ws_size,
                              hipStream_t stream)
{
    (void)in_sizes; (void)n_in; (void)out_size; (void)ws_size;
    // setup_inputs order: 0 input(empty), 1 h0, 2 c0, 3 yt, 4 eps,
    //                     5 W_ih, 6 W_hh, 7 b_ih, 8 b_hh, 9 W_proj, 10 b_proj
    const float* h0    = (const float*)d_in[1];
    const float* c0    = (const float*)d_in[2];
    const float* yt    = (const float*)d_in[3];
    const float* eps   = (const float*)d_in[4];
    const float* Wih   = (const float*)d_in[5];
    const float* Whh   = (const float*)d_in[6];
    const float* bih   = (const float*)d_in[7];
    const float* bhh   = (const float*)d_in[8];
    const float* Wproj = (const float*)d_in[9];
    const float* bproj = (const float*)d_in[10];

    // workspace layout (bytes)
    char* ws = (char*)d_ws;
    unsigned short* whh_b   = (unsigned short*)(ws);                       // 2 MiB
    unsigned short* wih_b   = (unsigned short*)(ws + 2097152);             // 256 KiB
    unsigned short* wproj_b = (unsigned short*)(ws + 2097152 + 262144);    // 128 KiB
    float* bias_o = (float*)(ws + 2097152 + 262144 + 131072);              // 8 KiB
    float* bpr_o  = (float*)(ws + 2097152 + 262144 + 131072 + 8192);       // 512 B

    arlstm_prep<<<512, 256, 0, stream>>>(Wih, Whh, bih, bhh, Wproj, bproj,
                                         whh_b, wih_b, wproj_b, bias_o, bpr_o);
    arlstm_main<<<16, 512, 0, stream>>>(h0, c0, yt, eps,
                                        wih_b, whh_b, wproj_b,
                                        bias_o, bpr_o, (float*)d_out);
}
